// AdaptiveLiquidNeuron_51694226374732
// MI455X (gfx1250) — matrix operand order repaired and checked by reading, not compiled, not run on hardware
//
#include <hip/hip_runtime.h>
#include <cstddef>
#include <cstdint>

// ---------------------------------------------------------------------------
// Problem constants (fixed by the reference)
// ---------------------------------------------------------------------------
static constexpr int kB = 16384;   // batch rows
static constexpr int kH = 1024;    // hidden dim
static constexpr float kEps = 1e-5f;

typedef __attribute__((ext_vector_type(16))) __bf16 v16bf;
typedef __attribute__((ext_vector_type(8)))  float  v8f;
typedef __attribute__((ext_vector_type(4)))  __bf16 v4bf;
typedef __attribute__((ext_vector_type(4)))  int    v4i;

// gfx1250 async global->LDS DMA (ASYNCcnt-tracked). Parameter types per the
// compiler's own diagnostic: (AS1 v4i* global_src, AS3 v4i* lds_dst, imm, imm).
#if __has_builtin(__builtin_amdgcn_global_load_async_to_lds_b128) && \
    __has_builtin(__builtin_amdgcn_s_wait_asynccnt)
#define HAVE_ASYNC_LDS 1
typedef __attribute__((address_space(1))) v4i* gptr_v4i;
typedef __attribute__((address_space(3))) v4i* lptr_v4i;
#else
#define HAVE_ASYNC_LDS 0
#endif

// ---------------------------------------------------------------------------
// fp32 -> bf16 streaming convert (float4 in, 4x bf16 out)
// ---------------------------------------------------------------------------
__global__ __launch_bounds__(256)
void cvt_f32_bf16_kernel(const float* __restrict__ in,
                         __bf16* __restrict__ out, int n4) {
    int i = blockIdx.x * blockDim.x + threadIdx.x;
    if (i < n4) {
        float4 f = reinterpret_cast<const float4*>(in)[i];
        v4bf o;
        o.x = (__bf16)f.x; o.y = (__bf16)f.y;
        o.z = (__bf16)f.z; o.w = (__bf16)f.w;
        reinterpret_cast<v4bf*>(out)[i] = o;
    }
}

// ---------------------------------------------------------------------------
// bf16 WMMA GEMM:  C[M,N] = act(A[M,K] @ W[N,K]^T + bias[N])
// Block 256 threads = 8 waves; block tile 128x128; wave tile 32(M)x64(N);
// TILE_K=64, double-buffered LDS filled by GLOBAL_LOAD_ASYNC_TO_LDS_B128,
// drained with s_wait_asynccnt; epilogue staged through LDS for coalesced
// vectorized global stores.
// ---------------------------------------------------------------------------
static constexpr int TILE_M = 128;
static constexpr int TILE_N = 128;
static constexpr int TILE_K = 64;
static constexpr int KPAD   = 8;                     // 16B row stagger, rows stay 16B aligned
static constexpr int TROW   = TILE_K + KPAD;         // 72 bf16 = 144 B pitch
static constexpr int TILE_BYTES = TILE_M * TROW * 2; // one tile buffer: 18432 B
static constexpr int SMEM_BYTES = 4 * TILE_BYTES;    // 2 tensors x 2 buffers = 73728 B (> 64KB f32 staging)

template <typename T>
__device__ inline void store4(T* p, float a, float b, float c, float d);
template <>
__device__ inline void store4<float>(float* p, float a, float b, float c, float d) {
    *reinterpret_cast<float4*>(p) = make_float4(a, b, c, d);
}
template <>
__device__ inline void store4<__bf16>(__bf16* p, float a, float b, float c, float d) {
    v4bf o;
    o.x = (__bf16)a; o.y = (__bf16)b; o.z = (__bf16)c; o.w = (__bf16)d;
    *reinterpret_cast<v4bf*>(p) = o;
}

template <bool RELU, typename OutT>
__global__ __launch_bounds__(256)
void gemm_bf16_wmma(const __bf16* __restrict__ A,
                    const __bf16* __restrict__ W,
                    const float*  __restrict__ bias,   // may be nullptr
                    OutT* __restrict__ C,
                    int M, int N, int K) {
    __shared__ __attribute__((aligned(16))) char smem[SMEM_BYTES];
    typedef __bf16 TileT[TILE_M][TROW];
    TileT* tA = reinterpret_cast<TileT*>(smem);                   // tA[buf][row][col]
    TileT* tB = reinterpret_cast<TileT*>(smem + 2 * TILE_BYTES);  // tB[buf][row][col]

    const int tid  = threadIdx.x;
    const int lane = tid & 31;
    const int wave = tid >> 5;
    const int waveM = wave & 3;   // 0..3 -> 32-row strip
    const int waveN = wave >> 2;  // 0..1 -> 64-col strip
    const int m0 = blockIdx.y * TILE_M;
    const int n0 = blockIdx.x * TILE_N;
    const int hl = lane & 15;

    v8f acc[2][4];
    {
        v8f z;
        #pragma unroll
        for (int q = 0; q < 8; ++q) z[q] = 0.0f;
        #pragma unroll
        for (int i = 0; i < 2; ++i)
            #pragma unroll
            for (int j = 0; j < 4; ++j) acc[i][j] = z;
    }

    // Per-thread staging assignment: 1024 x 16B chunks per tile, 4 per thread.
    int srow[4], scol[4];
    #pragma unroll
    for (int s = 0; s < 4; ++s) {
        int g = tid + s * 256;
        srow[s] = g >> 3;            // 0..127
        scol[s] = (g & 7) << 3;      // 0..56 (bf16 units, 16B aligned)
    }

    auto stage = [&](int buf, int kk) {
#if HAVE_ASYNC_LDS
        #pragma unroll
        for (int s = 0; s < 4; ++s) {
            const __bf16* ga = &A[(size_t)(m0 + srow[s]) * K + kk + scol[s]];
            const __bf16* gw = &W[(size_t)(n0 + srow[s]) * K + kk + scol[s]];
            __builtin_amdgcn_global_load_async_to_lds_b128(
                (gptr_v4i)ga, (lptr_v4i)&tA[buf][srow[s]][scol[s]], 0, 0);
            __builtin_amdgcn_global_load_async_to_lds_b128(
                (gptr_v4i)gw, (lptr_v4i)&tB[buf][srow[s]][scol[s]], 0, 0);
        }
#else
        uint4 ra[4], rb[4];
        #pragma unroll
        for (int s = 0; s < 4; ++s) {
            ra[s] = *reinterpret_cast<const uint4*>(&A[(size_t)(m0 + srow[s]) * K + kk + scol[s]]);
            rb[s] = *reinterpret_cast<const uint4*>(&W[(size_t)(n0 + srow[s]) * K + kk + scol[s]]);
        }
        #pragma unroll
        for (int s = 0; s < 4; ++s) {
            *reinterpret_cast<uint4*>(&tA[buf][srow[s]][scol[s]]) = ra[s];
            *reinterpret_cast<uint4*>(&tB[buf][srow[s]][scol[s]]) = rb[s];
        }
#endif
    };

    auto wait_stage = [&]() {
#if HAVE_ASYNC_LDS
        __builtin_amdgcn_s_wait_asynccnt(0);
#endif
        __syncthreads();
    };

    auto compute = [&](int buf) {
        #pragma unroll
        for (int ks = 0; ks < TILE_K; ks += 32) {
            // A 16x32 frag: lane<16 -> row M=lane, K{0..7,16..23};
            //               lane>=16 -> row M=lane-16, K{8..15,24..31}
            v16bf afrag[2];
            const int akoff = (lane < 16) ? 0 : 8;
            #pragma unroll
            for (int i = 0; i < 2; ++i) {
                const __bf16* p = &tA[buf][waveM * 32 + i * 16 + hl][ks];
                reinterpret_cast<uint4*>(&afrag[i])[0] =
                    *reinterpret_cast<const uint4*>(p + akoff);
                reinterpret_cast<uint4*>(&afrag[i])[1] =
                    *reinterpret_cast<const uint4*>(p + akoff + 16);
            }
            // B 32x16 frag: lane holds column N=lane%16; lane<16 -> K{0..7,16..23}, else K{8..15,24..31}
            v16bf bfrag[4];
            const int bkoff = (lane < 16) ? 0 : 8;
            #pragma unroll
            for (int j = 0; j < 4; ++j) {
                const __bf16* p = &tB[buf][waveN * 64 + j * 16 + hl][ks];
                reinterpret_cast<uint4*>(&bfrag[j])[0] =
                    *reinterpret_cast<const uint4*>(p + bkoff);
                reinterpret_cast<uint4*>(&bfrag[j])[1] =
                    *reinterpret_cast<const uint4*>(p + bkoff + 16);
            }
            #pragma unroll
            for (int i = 0; i < 2; ++i)
                #pragma unroll
                for (int j = 0; j < 4; ++j)
                    acc[i][j] = __builtin_amdgcn_wmma_f32_16x16x32_bf16(
                        false, afrag[i], false, bfrag[j],
                        (short)0, acc[i][j], false, false);
        }
    };

    // ---- double-buffered main loop: DMA next tile while WMMAs eat current ----
    int cur = 0;
    stage(0, 0);
    wait_stage();
    for (int kk = 0; kk < K; kk += TILE_K) {
        const bool more = (kk + TILE_K) < K;
        if (more) stage(cur ^ 1, kk + TILE_K);
        compute(cur);
        if (more) { wait_stage(); cur ^= 1; }
    }

    // ---- epilogue: stage accumulators to LDS (f32), then coalesced stores ----
    __syncthreads();                         // tile buffers dead; reuse as f32 staging
    float* sC = reinterpret_cast<float*>(smem);   // 128x128 f32 = 64KB
    #pragma unroll
    for (int i = 0; i < 2; ++i) {
        const int rbase = waveM * 32 + i * 16 + ((lane >> 4) << 3);
        #pragma unroll
        for (int j = 0; j < 4; ++j) {
            const int c = waveN * 64 + j * 16 + hl;
            #pragma unroll
            for (int v = 0; v < 8; ++v)
                sC[(rbase + v) * TILE_N + c] = acc[i][j][v];
        }
    }
    __syncthreads();

    const bool hasb = (bias != nullptr);
    #pragma unroll
    for (int s = 0; s < 16; ++s) {
        const int q  = tid + s * 256;        // 0..4095
        const int r  = q >> 5;               // row 0..127
        const int c4 = (q & 31) << 2;        // col 0,4,...,124
        float4 v = *reinterpret_cast<const float4*>(&sC[r * TILE_N + c4]);
        float4 bv = hasb ? *reinterpret_cast<const float4*>(&bias[n0 + c4])
                         : make_float4(0.f, 0.f, 0.f, 0.f);
        float x0 = v.x + bv.x, x1 = v.y + bv.y, x2 = v.z + bv.z, x3 = v.w + bv.w;
        if (RELU) {
            x0 = fmaxf(x0, 0.f); x1 = fmaxf(x1, 0.f);
            x2 = fmaxf(x2, 0.f); x3 = fmaxf(x3, 0.f);
        }
        store4<OutT>(&C[(size_t)(m0 + r) * N + n0 + c4], x0, x1, x2, x3);
    }
}

// ---------------------------------------------------------------------------
// Fused epilogue: dh = (-decay*h + (1+pa)*r + bias) / (tau*sigmoid(pa));
// then LayerNorm over the row.  One block per row; r aliases out (d_out):
// all reads of a row complete before any write in its block.
// ---------------------------------------------------------------------------
__global__ __launch_bounds__(256)
void liquid_epilogue_ln(const float* __restrict__ h,
                        const float* __restrict__ pa,
                        const float* __restrict__ r,
                        const float* __restrict__ bias,
                        const float* __restrict__ tau,
                        const float* __restrict__ decay,
                        const float* __restrict__ ln_w,
                        const float* __restrict__ ln_b,
                        float* __restrict__ out) {
    const int row = blockIdx.x;
    const int tid = threadIdx.x;
    const size_t base = (size_t)row * kH;

    float v[4];
    float sum = 0.0f, sq = 0.0f;
    #pragma unroll
    for (int k = 0; k < 4; ++k) {
        const int j = tid + k * 256;
        const float p   = pa[base + j];
        const float rv  = r[base + j];
        const float sig = 1.0f / (1.0f + __expf(-p));
        const float x   = (-decay[j] * h[base + j] + (1.0f + p) * rv + bias[j])
                          / (tau[j] * sig);
        v[k] = x;
        sum += x;
        sq  += x * x;
    }
    #pragma unroll
    for (int o = 16; o > 0; o >>= 1) {
        sum += __shfl_xor(sum, o, 32);
        sq  += __shfl_xor(sq,  o, 32);
    }
    __shared__ float s_sum[8], s_sq[8];
    if ((tid & 31) == 0) { s_sum[tid >> 5] = sum; s_sq[tid >> 5] = sq; }
    __syncthreads();
    float ts = 0.0f, tq = 0.0f;
    #pragma unroll
    for (int w = 0; w < 8; ++w) { ts += s_sum[w]; tq += s_sq[w]; }

    const float inv  = 1.0f / (float)kH;
    const float mu   = ts * inv;
    const float var  = tq * inv - mu * mu;
    const float rstd = rsqrtf(var + kEps);

    #pragma unroll
    for (int k = 0; k < 4; ++k) {
        const int j = tid + k * 256;
        out[base + j] = (v[k] - mu) * rstd * ln_w[j] + ln_b[j];
    }
}

// ---------------------------------------------------------------------------
// Host launcher
// ---------------------------------------------------------------------------
extern "C" void kernel_launch(void* const* d_in, const int* in_sizes, int n_in,
                              void* d_out, int out_size, void* d_ws, size_t ws_size,
                              hipStream_t stream) {
    (void)in_sizes; (void)n_in; (void)out_size; (void)ws_size;

    const float* h     = (const float*)d_in[1];
    const float* e     = (const float*)d_in[2];
    const float* W_rec = (const float*)d_in[3];
    const float* bias  = (const float*)d_in[4];
    const float* tau   = (const float*)d_in[5];
    const float* decay = (const float*)d_in[6];
    const float* ln_w  = (const float*)d_in[7];
    const float* ln_b  = (const float*)d_in[8];
    const float* ce_w1 = (const float*)d_in[9];
    const float* ce_b1 = (const float*)d_in[10];
    const float* ce_w2 = (const float*)d_in[11];
    const float* ce_b2 = (const float*)d_in[12];
    const float* pm_w  = (const float*)d_in[13];
    const float* pm_b  = (const float*)d_in[14];
    float* out = (float*)d_out;

    // Workspace (136 MB): bufA/bufB = B*H bf16 ping-pong, pa = B*H f32,
    // four H*H bf16 weight copies.
    char* ws = (char*)d_ws;
    const size_t szBH_bf = (size_t)kB * kH * sizeof(__bf16);
    const size_t szBH_f  = (size_t)kB * kH * sizeof(float);
    const size_t szHH_bf = (size_t)kH * kH * sizeof(__bf16);

    __bf16* bufA = (__bf16*)(ws);
    __bf16* bufB = (__bf16*)(ws + szBH_bf);
    float*  pa   = (float*) (ws + 2 * szBH_bf);
    __bf16* w1b  = (__bf16*)(ws + 2 * szBH_bf + szBH_f);
    __bf16* w2b  = (__bf16*)(ws + 2 * szBH_bf + szBH_f + 1 * szHH_bf);
    __bf16* pmb  = (__bf16*)(ws + 2 * szBH_bf + szBH_f + 2 * szHH_bf);
    __bf16* wrb  = (__bf16*)(ws + 2 * szBH_bf + szBH_f + 3 * szHH_bf);

    const int nBH4 = (kB * kH) / 4;
    const int nHH4 = (kH * kH) / 4;
    const dim3 cvtBH((nBH4 + 255) / 256), cvtHH((nHH4 + 255) / 256), tb(256);

    cvt_f32_bf16_kernel<<<cvtBH, tb, 0, stream>>>(h,     bufA, nBH4);
    cvt_f32_bf16_kernel<<<cvtHH, tb, 0, stream>>>(ce_w1, w1b,  nHH4);
    cvt_f32_bf16_kernel<<<cvtHH, tb, 0, stream>>>(ce_w2, w2b,  nHH4);
    cvt_f32_bf16_kernel<<<cvtHH, tb, 0, stream>>>(pm_w,  pmb,  nHH4);
    cvt_f32_bf16_kernel<<<cvtHH, tb, 0, stream>>>(W_rec, wrb,  nHH4);

    const dim3 gg(kH / TILE_N, kB / TILE_M);   // (8, 128)

    // G1: a1 = ReLU(h @ ce_w1^T + ce_b1)  (bf16)
    gemm_bf16_wmma<true,  __bf16><<<gg, tb, 0, stream>>>(bufA, w1b, ce_b1, bufB, kB, kH, kH);
    // G2: ctx = a1 @ ce_w2^T + ce_b2      (bf16, into bufA)
    gemm_bf16_wmma<false, __bf16><<<gg, tb, 0, stream>>>(bufB, w2b, ce_b2, bufA, kB, kH, kH);
    // G3: pa = ctx @ pm_w^T + pm_b        (f32)
    gemm_bf16_wmma<false, float ><<<gg, tb, 0, stream>>>(bufA, pmb, pm_b, pa, kB, kH, kH);
    // e -> bf16 (bufB free after G2)
    cvt_f32_bf16_kernel<<<cvtBH, tb, 0, stream>>>(e, bufB, nBH4);
    // G4: r = e @ W_rec^T                 (f32, d_out used as scratch)
    gemm_bf16_wmma<false, float ><<<gg, tb, 0, stream>>>(bufB, wrb, nullptr, out, kB, kH, kH);

    // fused modulation + LayerNorm (reads r from d_out, overwrites d_out)
    liquid_epilogue_ln<<<dim3(kB), tb, 0, stream>>>(h, pa, out, bias, tau, decay,
                                                    ln_w, ln_b, out);
}
